// MultiHeadedAttention_42683384988075
// MI455X (gfx1250) — matrix operand order repaired and checked by reading, not compiled, not run on hardware
//
#include <hip/hip_runtime.h>

// ---------------------------------------------------------------------------
// Multi-head attention for gfx1250 (MI455X), bf16 WMMA path.
// B=8, S=2048, D=512, H=8, DK=64.
// ---------------------------------------------------------------------------

typedef __attribute__((ext_vector_type(16))) __bf16          v16bf;
typedef __attribute__((ext_vector_type(8)))  float           v8f;
typedef __attribute__((ext_vector_type(4)))  float           v4f;
typedef __attribute__((ext_vector_type(8)))  unsigned short  v8u16;
typedef __attribute__((ext_vector_type(16))) unsigned short  v16u16;

#define DEVINL __device__ __forceinline__

static constexpr int BATCH = 8;
static constexpr int SEQ   = 2048;
static constexpr int DMODEL= 512;
static constexpr int HEADS = 8;
static constexpr int DK    = 64;

DEVINL unsigned short f2bf(float f) {
  unsigned int u = __float_as_uint(f);
  u += 0x7FFFu + ((u >> 16) & 1u);          // round-to-nearest-even
  return (unsigned short)(u >> 16);
}

DEVINL v8f wmma_bf16(v16bf a, v16bf b, v8f c) {
  // (neg_a, A, neg_b, B, c_mod, C, reuse_a, reuse_b)
  return __builtin_amdgcn_wmma_f32_16x16x32_bf16(false, a, false, b,
                                                 (short)0, c, false, false);
}

// A-matrix 16x32 bf16 fragment from row-major (row=M, contiguous K) storage.
// ISA layout: lane half h holds K = h*8..h*8+7 (V0..V3) and 16+h*8.. (V4..V7).
DEVINL v16bf lda_frag_u16(const unsigned short* base, int ld) {
  const int lane = threadIdx.x & 31;
  const int h = lane >> 4, r = lane & 15;
  v8u16 lo = *(const v8u16*)(base + (size_t)r * ld + h * 8);
  v8u16 hi = *(const v8u16*)(base + (size_t)r * ld + 16 + h * 8);
  v16u16 t;
#pragma unroll
  for (int i = 0; i < 8; ++i) { t[i] = lo[i]; t[8 + i] = hi[i]; }
  return __builtin_bit_cast(v16bf, t);
}

// B-matrix 32x16 bf16 fragment; base is B^T row-major (row = N column of B,
// contiguous K). ISA layout: lane half h holds K = h*8..h*8+7 and 16+h*8..
DEVINL v16bf ldb_frag_u16(const unsigned short* base, int ld) {
  const int lane = threadIdx.x & 31;
  const int h = lane >> 4, r = lane & 15;
  v8u16 lo = *(const v8u16*)(base + (size_t)r * ld + h * 8);
  v8u16 hi = *(const v8u16*)(base + (size_t)r * ld + 16 + h * 8);
  v16u16 t;
#pragma unroll
  for (int i = 0; i < 8; ++i) { t[i] = lo[i]; t[8 + i] = hi[i]; }
  return __builtin_bit_cast(v16bf, t);
}

// A fragment built from f32 data (LDS softmax-P tile), converting to bf16.
DEVINL v16bf lda_frag_f32(const float* base, int ld) {
  const int lane = threadIdx.x & 31;
  const int h = lane >> 4, r = lane & 15;
  const float* p0 = base + (size_t)r * ld + h * 8;
  const float* p1 = base + (size_t)r * ld + 16 + h * 8;
  v4f a0 = *(const v4f*)p0, a1 = *(const v4f*)(p0 + 4);
  v4f b0 = *(const v4f*)p1, b1 = *(const v4f*)(p1 + 4);
  v16u16 t;
#pragma unroll
  for (int i = 0; i < 4; ++i) {
    t[i]      = f2bf(a0[i]);
    t[4 + i]  = f2bf(a1[i]);
    t[8 + i]  = f2bf(b0[i]);
    t[12 + i] = f2bf(b1[i]);
  }
  return __builtin_bit_cast(v16bf, t);
}

// ---------------------------------------------------------------------------
// GEMM + bias:  out[m,n] = (sum_k A[m,k] * W[n,k] + bias[n]) * scale
// M=16384, N=K=512. 128 threads (4 waves) per block, 64x64 tile,
// LDS-staged 64x32 bf16 operand tiles, 4 x v_wmma_f32_16x16x32_bf16 per step.
// AMODE: 0 = A is f32 (convert while staging), 1 = A is bf16 (u16).
// OMODE: 0 = bf16 row-major [M,N], 1 = bf16 per-head transposed [B,H,DK,S],
//        2 = f32 row-major (final output).
// ---------------------------------------------------------------------------
template <int AMODE, int OMODE>
__global__ __launch_bounds__(128) void gemm_bias_kernel(
    const void* __restrict__ Aptr, const float* __restrict__ W,
    const float* __restrict__ bias, void* __restrict__ out, float scale) {
  constexpr int KD = DMODEL, N = DMODEL;
  __shared__ unsigned short sA[64 * 32];
  __shared__ unsigned short sB[64 * 32];

  const int n0 = blockIdx.x * 64;
  const int m0 = blockIdx.y * 64;
  const int wave = threadIdx.x >> 5;
  const int lane = threadIdx.x & 31;
  const int wm = wave >> 1, wn = wave & 1;

  const float*          Af = (const float*)Aptr;
  const unsigned short* Ab = (const unsigned short*)Aptr;

  v8f acc[2][2] = {};

  for (int k0 = 0; k0 < KD; k0 += 32) {
    // --- cooperative staging: 256 16B chunks per tile, 2 per thread ---
    for (int c = threadIdx.x; c < 256; c += 128) {
      const int row = c >> 2, ch = c & 3;
      if (AMODE == 0) {
        const float* p = Af + (size_t)(m0 + row) * KD + k0 + ch * 8;
        v4f x0 = *(const v4f*)p, x1 = *(const v4f*)(p + 4);
        v8u16 t;
#pragma unroll
        for (int i = 0; i < 4; ++i) { t[i] = f2bf(x0[i]); t[4 + i] = f2bf(x1[i]); }
        *(v8u16*)&sA[row * 32 + ch * 8] = t;
      } else {
        *(v8u16*)&sA[row * 32 + ch * 8] =
            *(const v8u16*)(Ab + (size_t)(m0 + row) * KD + k0 + ch * 8);
      }
      const float* pw = W + (size_t)(n0 + row) * KD + k0 + ch * 8;
      v4f w0 = *(const v4f*)pw, w1 = *(const v4f*)(pw + 4);
      v8u16 tw;
#pragma unroll
      for (int i = 0; i < 4; ++i) { tw[i] = f2bf(w0[i]); tw[4 + i] = f2bf(w1[i]); }
      *(v8u16*)&sB[row * 32 + ch * 8] = tw;
    }
    __syncthreads();

    v16bf a0 = lda_frag_u16(&sA[(wm * 32 + 0)  * 32], 32);
    v16bf a1 = lda_frag_u16(&sA[(wm * 32 + 16) * 32], 32);
    v16bf b0 = ldb_frag_u16(&sB[(wn * 32 + 0)  * 32], 32);
    v16bf b1 = ldb_frag_u16(&sB[(wn * 32 + 16) * 32], 32);
    acc[0][0] = wmma_bf16(a0, b0, acc[0][0]);
    acc[0][1] = wmma_bf16(a0, b1, acc[0][1]);
    acc[1][0] = wmma_bf16(a1, b0, acc[1][0]);
    acc[1][1] = wmma_bf16(a1, b1, acc[1][1]);
    __syncthreads();
  }

  // --- epilogue: C layout lane(h,r), VGPR i -> (M = h*8+i, N = r) ---
  const int h = lane >> 4, r = lane & 15;
#pragma unroll
  for (int mt = 0; mt < 2; ++mt) {
#pragma unroll
    for (int nt = 0; nt < 2; ++nt) {
      const int n  = n0 + wn * 32 + nt * 16 + r;
      const float bn = bias[n];
#pragma unroll
      for (int i = 0; i < 8; ++i) {
        const int m = m0 + wm * 32 + mt * 16 + h * 8 + i;
        const float val = (acc[mt][nt][i] + bn) * scale;
        if (OMODE == 2) {
          ((float*)out)[(size_t)m * N + n] = val;
        } else if (OMODE == 0) {
          ((unsigned short*)out)[(size_t)m * N + n] = f2bf(val);
        } else {  // per-head transposed: [B, H, DK, S]
          const int bb = m >> 11, s  = m & (SEQ - 1);
          const int hh = n >> 6,  dk = n & (DK - 1);
          ((unsigned short*)out)[(((size_t)bb * HEADS + hh) * DK + dk) * SEQ + s] =
              f2bf(val);
        }
      }
    }
  }
}

// ---------------------------------------------------------------------------
// Flash attention: grid (S/64, H, B), 128 threads = 4 waves, each wave owns
// 16 query rows. Online softmax over KV tiles of 32. Scale is pre-folded
// into Q at projection time.
// ---------------------------------------------------------------------------
__global__ __launch_bounds__(128) void attn_kernel(
    const unsigned short* __restrict__ Qp,   // [B*S, D] bf16 (pre-scaled)
    const unsigned short* __restrict__ Kp,   // [B*S, D] bf16
    const unsigned short* __restrict__ Vt,   // [B, H, DK, S] bf16
    unsigned short* __restrict__ Xo) {       // [B*S, D] bf16
  __shared__ unsigned short sK[32 * 64];     // [kv 0..31][dk 0..63]
  __shared__ unsigned short sV[64 * 32];     // [dk 0..63][kv 0..31]
  __shared__ float          sP[4][16 * 32];  // per-wave P tile, f32

  const int qblk = blockIdx.x;
  const int h    = blockIdx.y;
  const int b    = blockIdx.z;
  const int wave = threadIdx.x >> 5;
  const int lane = threadIdx.x & 31;
  const int hh = lane >> 4, r = lane & 15;
  const int q0 = qblk * 64 + wave * 16;

  // persistent Q fragments (K = 0..31 and 32..63 of DK)
  const unsigned short* Qb = Qp + ((size_t)(b * SEQ + q0)) * DMODEL + h * DK;
  const v16bf qf0 = lda_frag_u16(Qb, DMODEL);
  const v16bf qf1 = lda_frag_u16(Qb + 32, DMODEL);

  v8f o0 = {}, o1 = {}, o2 = {}, o3 = {};
  float m_i[8], l_i[8];
#pragma unroll
  for (int i = 0; i < 8; ++i) { m_i[i] = -3.0e38f; l_i[i] = 0.0f; }

  for (int kv0 = 0; kv0 < SEQ; kv0 += 32) {
    // --- stage K tile: 32 rows x 128B ---
    for (int c = threadIdx.x; c < 256; c += 128) {
      const int row = c >> 3, ch = c & 7;
      *(v8u16*)&sK[row * 64 + ch * 8] =
          *(const v8u16*)(Kp + ((size_t)(b * SEQ + kv0 + row)) * DMODEL +
                          h * DK + ch * 8);
    }
    // --- stage V^T tile: 64 rows x 64B ---
    for (int c = threadIdx.x; c < 256; c += 128) {
      const int row = c >> 2, ch = c & 3;
      *(v8u16*)&sV[row * 32 + ch * 8] =
          *(const v8u16*)(Vt + (((size_t)(b * HEADS + h)) * DK + row) * SEQ +
                          kv0 + ch * 8);
    }
    // prefetch next KV tile (global_prefetch_b8)
    if (kv0 + 32 < SEQ) {
      const int prow = threadIdx.x & 31;
      __builtin_prefetch(
          Kp + ((size_t)(b * SEQ + kv0 + 32 + prow)) * DMODEL + h * DK, 0, 1);
    }
    __syncthreads();

    // --- scores: M=16 q, N=32 kv, K=64 dk -> 4 WMMAs ---
    v8f s0 = {}, s1 = {};
    s0 = wmma_bf16(qf0, ldb_frag_u16(&sK[0 * 16 * 64 + 0],  64), s0);
    s0 = wmma_bf16(qf1, ldb_frag_u16(&sK[0 * 16 * 64 + 32], 64), s0);
    s1 = wmma_bf16(qf0, ldb_frag_u16(&sK[1 * 16 * 64 + 0],  64), s1);
    s1 = wmma_bf16(qf1, ldb_frag_u16(&sK[1 * 16 * 64 + 32], 64), s1);

    // --- online softmax; row reductions stay within 16-lane halves ---
    float* Pw = sP[wave];
#pragma unroll
    for (int i = 0; i < 8; ++i) {
      float pm = fmaxf(s0[i], s1[i]);
      pm = fmaxf(pm, __shfl_xor(pm, 1, 32));
      pm = fmaxf(pm, __shfl_xor(pm, 2, 32));
      pm = fmaxf(pm, __shfl_xor(pm, 4, 32));
      pm = fmaxf(pm, __shfl_xor(pm, 8, 32));
      const float mn    = fmaxf(m_i[i], pm);
      const float alpha = __expf(m_i[i] - mn);
      const float p0    = __expf(s0[i] - mn);
      const float p1    = __expf(s1[i] - mn);
      float ps = p0 + p1;
      ps += __shfl_xor(ps, 1, 32);
      ps += __shfl_xor(ps, 2, 32);
      ps += __shfl_xor(ps, 4, 32);
      ps += __shfl_xor(ps, 8, 32);
      l_i[i] = l_i[i] * alpha + ps;
      m_i[i] = mn;
      o0[i] *= alpha; o1[i] *= alpha; o2[i] *= alpha; o3[i] *= alpha;
      Pw[(hh * 8 + i) * 32 + r]      = p0;
      Pw[(hh * 8 + i) * 32 + 16 + r] = p1;
    }
    __syncthreads();

    // --- PV: M=16 q, N=64 dk, K=32 kv -> 4 WMMAs ---
    const v16bf pf = lda_frag_f32(Pw, 32);
    o0 = wmma_bf16(pf, ldb_frag_u16(&sV[0  * 32], 32), o0);
    o1 = wmma_bf16(pf, ldb_frag_u16(&sV[16 * 32], 32), o1);
    o2 = wmma_bf16(pf, ldb_frag_u16(&sV[32 * 32], 32), o2);
    o3 = wmma_bf16(pf, ldb_frag_u16(&sV[48 * 32], 32), o3);
    __syncthreads();
  }

  // --- epilogue: divide by l and store bf16 ---
#pragma unroll
  for (int i = 0; i < 8; ++i) {
    const float inv = 1.0f / l_i[i];
    const int m = q0 + hh * 8 + i;
    unsigned short* p = Xo + ((size_t)(b * SEQ + m)) * DMODEL + h * DK;
    p[r]      = f2bf(o0[i] * inv);
    p[16 + r] = f2bf(o1[i] * inv);
    p[32 + r] = f2bf(o2[i] * inv);
    p[48 + r] = f2bf(o3[i] * inv);
  }
}

// ---------------------------------------------------------------------------
extern "C" void kernel_launch(void* const* d_in, const int* in_sizes, int n_in,
                              void* d_out, int out_size, void* d_ws,
                              size_t ws_size, hipStream_t stream) {
  (void)in_sizes; (void)n_in; (void)out_size; (void)ws_size;

  const float* query = (const float*)d_in[0];
  const float* key   = (const float*)d_in[1];
  const float* value = (const float*)d_in[2];
  const float* wq = (const float*)d_in[3];
  const float* bq = (const float*)d_in[4];
  const float* wk = (const float*)d_in[5];
  const float* bk = (const float*)d_in[6];
  const float* wv = (const float*)d_in[7];
  const float* bv = (const float*)d_in[8];
  const float* wo = (const float*)d_in[9];
  const float* bo = (const float*)d_in[10];

  const size_t NE = (size_t)BATCH * SEQ * DMODEL;  // 8388608 elements
  unsigned short* Qp = (unsigned short*)d_ws;      // bf16 [B*S, D], pre-scaled
  unsigned short* Kp = Qp + NE;                    // bf16 [B*S, D]
  unsigned short* Vt = Kp + NE;                    // bf16 [B, H, DK, S]
  unsigned short* Xa = Vt + NE;                    // bf16 [B*S, D]

  const dim3 blk(128);
  const dim3 gg(DMODEL / 64, (BATCH * SEQ) / 64);  // (8, 256)
  const float qscale = 0.125f;                     // 1/sqrt(DK)

  gemm_bias_kernel<0, 0><<<gg, blk, 0, stream>>>(query, wq, bq, Qp, qscale);
  gemm_bias_kernel<0, 0><<<gg, blk, 0, stream>>>(key,   wk, bk, Kp, 1.0f);
  gemm_bias_kernel<0, 1><<<gg, blk, 0, stream>>>(value, wv, bv, Vt, 1.0f);

  attn_kernel<<<dim3(SEQ / 64, HEADS, BATCH), blk, 0, stream>>>(Qp, Kp, Vt, Xa);

  gemm_bias_kernel<1, 2><<<gg, blk, 0, stream>>>(Xa, wo, bo, d_out, 1.0f);
}
